// default_gnn_74887049773805
// MI455X (gfx1250) — matrix operand order repaired and checked by reading, not compiled, not run on hardware
//
#include <hip/hip_runtime.h>
#include <hip/hip_bf16.h>

typedef __attribute__((ext_vector_type(16))) _Float16 v16h;
typedef __attribute__((ext_vector_type(8)))  float    v8f;

// ---------------------------------------------------------------------------
// Heavy path (f32, memory-bound, L2-resident working set):
// ---------------------------------------------------------------------------

__global__ void deg_kernel(const int* __restrict__ src, float* __restrict__ deg, int E) {
    int e = blockIdx.x * blockDim.x + threadIdx.x;
    if (e < E) atomicAdd(&deg[src[e]], 1.0f);
}

__global__ void dis_kernel(const float* __restrict__ deg, float* __restrict__ dis, int n) {
    int i = blockIdx.x * blockDim.x + threadIdx.x;
    if (i < n) {
        float d = deg[i];
        dis[i] = (d > 0.0f) ? rsqrtf(fmaxf(d, 1.0f)) : 0.0f;
    }
}

// T1 scatter (E x 64 atomics into L2-resident 25.6MB buffer) fused with the
// s1 = sum(T1) edge reduction (LDS partials -> 64 global atomics per block).
// 16 threads per edge, one float4 (4 channels) per thread -> coalesced 256B
// gathers of x rows.
__global__ void scatter_t1_kernel(const float4* __restrict__ x4,
                                  const int* __restrict__ src, const int* __restrict__ dst,
                                  const float* __restrict__ dis,
                                  float* __restrict__ T1, float* __restrict__ s1g, int E) {
    __shared__ float sh[64];
    int tid = threadIdx.x;
    if (tid < 64) sh[tid] = 0.0f;
    __syncthreads();
    int g = blockIdx.x * blockDim.x + tid;   // E*16 = 25.6M < 2^31
    int e = g >> 4;
    int part = g & 15;
    if (e < E) {
        int s = src[e], d = dst[e];
        float f = -dis[s] * dis[d];                  // T1 = -segsum(w * x[src], dst)
        float4 xv = x4[(long)s * 16 + part];
        float vx = f * xv.x, vy = f * xv.y, vz = f * xv.z, vw = f * xv.w;
        float* tp = &T1[(long)d * 64 + part * 4];
        atomicAdd(tp + 0, vx); atomicAdd(tp + 1, vy);
        atomicAdd(tp + 2, vz); atomicAdd(tp + 3, vw);
        int c = part * 4;
        atomicAdd(&sh[c + 0], vx); atomicAdd(&sh[c + 1], vy);
        atomicAdd(&sh[c + 2], vz); atomicAdd(&sh[c + 3], vw);
    }
    __syncthreads();
    if (tid < 64) atomicAdd(&s1g[tid], sh[tid]);
}

// s2acc = -2 * sum_e w_e * T1[src_e]   (gather over L2-resident T1)
__global__ void gather_s2_kernel(const float4* __restrict__ T14,
                                 const int* __restrict__ src, const int* __restrict__ dst,
                                 const float* __restrict__ dis,
                                 float* __restrict__ s2g, int E) {
    __shared__ float sh[64];
    int tid = threadIdx.x;
    if (tid < 64) sh[tid] = 0.0f;
    __syncthreads();
    int g = blockIdx.x * blockDim.x + tid;
    int e = g >> 4;
    int part = g & 15;
    if (e < E) {
        int s = src[e], d = dst[e];
        float f = -2.0f * dis[s] * dis[d];
        float4 tv = T14[(long)s * 16 + part];
        int c = part * 4;
        atomicAdd(&sh[c + 0], f * tv.x); atomicAdd(&sh[c + 1], f * tv.y);
        atomicAdd(&sh[c + 2], f * tv.z); atomicAdd(&sh[c + 3], f * tv.w);
    }
    __syncthreads();
    if (tid < 64) atomicAdd(&s2g[tid], sh[tid]);
}

// s0 = column sums of x
__global__ void colsum_kernel(const float4* __restrict__ x4, float* __restrict__ s0g, int n) {
    __shared__ float sh[64];
    int tid = threadIdx.x;
    if (tid < 64) sh[tid] = 0.0f;
    __syncthreads();
    int g = blockIdx.x * blockDim.x + tid;
    int nn = g >> 4;
    int part = g & 15;
    if (nn < n) {
        float4 v = x4[(long)nn * 16 + part];
        int c = part * 4;
        atomicAdd(&sh[c + 0], v.x); atomicAdd(&sh[c + 1], v.y);
        atomicAdd(&sh[c + 2], v.z); atomicAdd(&sh[c + 3], v.w);
    }
    __syncthreads();
    if (tid < 64) atomicAdd(&s0g[tid], sh[tid]);
}

// ---------------------------------------------------------------------------
// Tail: row-vector GEMMs via v_wmma_f32_16x16x32_f16 (single wave, full EXEC).
// out[n] = acc * scale + bias[n].  K % 32 == 0, NCOLS % 16 == 0.
// A fragments are precomputed once (masked, vectorized LDS loads — no
// divergent exec blocks) and reused across all N-tiles.
// ---------------------------------------------------------------------------
template<int K, int NCOLS, int TRANSB, int LD>
__device__ inline void row_gemm_wmma(const float* A,
                                     const float* __restrict__ Bg,
                                     const float* __restrict__ bias,
                                     float scale, float* out) {
    constexpr int NK = K / 32;
    int lane = threadIdx.x & 31;
    int m    = lane & 15;                 // A row this lane feeds (only M=0 is live)
    int aoff = (lane < 16) ? 0 : 8;       // ISA 16-bit A 16x32 K-chunk select
    int boff = (lane < 16) ? 0 : 8;       // B: lanes 0-15 hold K 0..7,16..23; 16-31 K 8..15,24..31
    int n    = lane & 15;                 // B/D column
    float amask = (m == 0) ? 1.0f : 0.0f; // arithmetic mask: no exec divergence

    v16h afrag[NK];
    #pragma unroll
    for (int kk = 0; kk < NK; ++kk) {
        int kb = kk * 32;
        float4 p0 = *(const float4*)&A[kb + aoff];
        float4 p1 = *(const float4*)&A[kb + aoff + 4];
        float4 p2 = *(const float4*)&A[kb + 16 + aoff];
        float4 p3 = *(const float4*)&A[kb + 16 + aoff + 4];
        afrag[kk][0]  = (_Float16)(p0.x * amask);
        afrag[kk][1]  = (_Float16)(p0.y * amask);
        afrag[kk][2]  = (_Float16)(p0.z * amask);
        afrag[kk][3]  = (_Float16)(p0.w * amask);
        afrag[kk][4]  = (_Float16)(p1.x * amask);
        afrag[kk][5]  = (_Float16)(p1.y * amask);
        afrag[kk][6]  = (_Float16)(p1.z * amask);
        afrag[kk][7]  = (_Float16)(p1.w * amask);
        afrag[kk][8]  = (_Float16)(p2.x * amask);
        afrag[kk][9]  = (_Float16)(p2.y * amask);
        afrag[kk][10] = (_Float16)(p2.z * amask);
        afrag[kk][11] = (_Float16)(p2.w * amask);
        afrag[kk][12] = (_Float16)(p3.x * amask);
        afrag[kk][13] = (_Float16)(p3.y * amask);
        afrag[kk][14] = (_Float16)(p3.z * amask);
        afrag[kk][15] = (_Float16)(p3.w * amask);
    }

    for (int t = 0; t < NCOLS; t += 16) {
        v8f acc = {};
        #pragma unroll
        for (int kk = 0; kk < NK; ++kk) {
            int kb = kk * 32;
            v16h b;
            #pragma unroll
            for (int i = 0; i < 16; ++i) {
                int k = kb + boff + (i < 8 ? i : i + 8);
                int col = t + n;
                float v = TRANSB ? Bg[(long)col * LD + k] : Bg[(long)k * LD + col];
                b[i] = (_Float16)v;
            }
            acc = __builtin_amdgcn_wmma_f32_16x16x32_f16(
                false, afrag[kk], false, b, (short)0, acc, false, false);
        }
        // D layout: VGPR0 of lanes 0-15 holds M=0, N=lane
        if (lane < 16) out[t + lane] = acc[0] * scale + bias[t + lane];
    }
}

__global__ void tail_kernel(const float* __restrict__ s0, const float* __restrict__ s1,
                            const float* __restrict__ s2acc,
                            const float* __restrict__ Wcheb, const float* __restrict__ bcheb,
                            const float* __restrict__ Wd, const float* __restrict__ bd,
                            const float* __restrict__ Wf, const float* __restrict__ bf,
                            float* __restrict__ y, float invN) {
    __shared__ __align__(16) float Abuf[256];
    __shared__ __align__(16) float Obuf[256];
    int lane = threadIdx.x;
    // A = [s0 | s1 | s2] (K=192), matching W_cheb row index r = k*64 + i
    for (int i = lane; i < 192; i += 32) {
        int k = i >> 6, c = i & 63;
        Abuf[i] = (k == 0) ? s0[c] : (k == 1) ? s1[c] : (s2acc[c] - s0[c]);
    }
    __syncthreads();
    // pooled[64] = (A @ Wcheb)/N + b_cheb
    row_gemm_wmma<192, 64, 0, 64>(Abuf, Wcheb, bcheb, invN, Obuf);
    __syncthreads();
    for (int i = lane; i < 64; i += 32) Abuf[i] = Obuf[i];
    __syncthreads();
    // h[256] = pooled @ Wd.T + b_dense   (Wd is [256,64] row-major -> transB)
    row_gemm_wmma<64, 256, 1, 64>(Abuf, Wd, bd, 1.0f, Obuf);
    __syncthreads();
    for (int i = lane; i < 256; i += 32) Abuf[i] = Obuf[i];
    __syncthreads();
    // y[128] = h @ Wf.T + b_final       (Wf is [128,256] row-major -> transB)
    row_gemm_wmma<256, 128, 1, 256>(Abuf, Wf, bf, 1.0f, y);
}

// ---------------------------------------------------------------------------

extern "C" void kernel_launch(void* const* d_in, const int* in_sizes, int n_in,
                              void* d_out, int out_size, void* d_ws, size_t ws_size,
                              hipStream_t stream) {
    const float* x     = (const float*)d_in[0];
    const int*   ei    = (const int*)  d_in[1];
    const float* Wcheb = (const float*)d_in[2];
    const float* bcheb = (const float*)d_in[3];
    const float* Wd    = (const float*)d_in[4];
    const float* bd    = (const float*)d_in[5];
    const float* Wf    = (const float*)d_in[6];
    const float* bf    = (const float*)d_in[7];

    int N = in_sizes[0] / 64;
    int E = in_sizes[1] / 2;

    float* ws  = (float*)d_ws;
    float* deg = ws;                          // [N]
    float* dis = ws + (size_t)N;              // [N]
    float* T1  = ws + (size_t)2 * N;          // [N*64]  (25.6 MB, L2-resident)
    float* s0g = ws + (size_t)66 * N;         // [64]
    float* s1g = s0g + 64;                    // [64]
    float* s2g = s0g + 128;                   // [64]

    // Re-zero accumulators every call (graph-capture-safe memset node).
    size_t zero_bytes = ((size_t)66 * N + 192) * sizeof(float);
    hipMemsetAsync(d_ws, 0, zero_bytes, stream);

    const int* src = ei;
    const int* dst = ei + E;

    deg_kernel<<<(E + 255) / 256, 256, 0, stream>>>(src, deg, E);
    dis_kernel<<<(N + 255) / 256, 256, 0, stream>>>(deg, dis, N);
    scatter_t1_kernel<<<(E * 16 + 255) / 256, 256, 0, stream>>>(
        (const float4*)x, src, dst, dis, T1, s1g, E);
    gather_s2_kernel<<<(E * 16 + 255) / 256, 256, 0, stream>>>(
        (const float4*)T1, src, dst, dis, s2g, E);
    colsum_kernel<<<(N * 16 + 255) / 256, 256, 0, stream>>>(
        (const float4*)x, s0g, N);
    tail_kernel<<<1, 32, 0, stream>>>(s0g, s1g, s2g, Wcheb, bcheb, Wd, bd, Wf, bf,
                                      (float*)d_out, 1.0f / (float)N);
}
